// ClassCaps_44899588113033
// MI455X (gfx1250) — matrix operand order repaired and checked by reading, not compiled, not run on hardware
//
#include <hip/hip_runtime.h>
#include <hip/hip_fp16.h>

// CapsNet dynamic routing for MI455X (gfx1250, wave32, WMMA).
// B=32 batch, K=1024 capsules, C=64 classes, O=32 out-dim, I=32 in-dim.

typedef __attribute__((ext_vector_type(16))) _Float16 v16h;
typedef __attribute__((ext_vector_type(8)))  _Float16 v8h;
typedef __attribute__((ext_vector_type(8)))  float    v8f;

#define Bsz   32
#define Kc    1024
#define Cc    64
#define Oc    32
#define Ic    32
#define MROWS 2048   // C*O rows of the per-k GEMM

__device__ __forceinline__ v16h cvt16(float4 a, float4 b, float4 c, float4 d) {
  v16h f;
  f[0]=(_Float16)a.x;  f[1]=(_Float16)a.y;  f[2]=(_Float16)a.z;  f[3]=(_Float16)a.w;
  f[4]=(_Float16)b.x;  f[5]=(_Float16)b.y;  f[6]=(_Float16)b.z;  f[7]=(_Float16)b.w;
  f[8]=(_Float16)c.x;  f[9]=(_Float16)c.y;  f[10]=(_Float16)c.z; f[11]=(_Float16)c.w;
  f[12]=(_Float16)d.x; f[13]=(_Float16)d.y; f[14]=(_Float16)d.z; f[15]=(_Float16)d.w;
  return f;
}

// -------------------------------------------------------------------------
// Kernel 1: u_hat[b,k,r] = sum_i W[k,r,i] * x[b,k,i]   (r = c*32+o, M=2048)
// One block per k (8 waves). Each wave: 16 M-tiles x 2 N-tiles of
// v_wmma_f32_16x16x32_f16 (K=I=32 in a single WMMA). D stored as f16
// (128 MB total -> fits the 192 MB L2 for the routing sweeps).
// -------------------------------------------------------------------------
__global__ __launch_bounds__(256)
void caps_uhat_wmma(const float* __restrict__ W, const float* __restrict__ X,
                    _Float16* __restrict__ uhat)
{
  const int k    = blockIdx.x;
  const int lane = threadIdx.x & 31;
  const int wv   = threadIdx.x >> 5;

  // B fragments (32x16 f16): lanes 0-15 = column `lane`, K=0..15 in elems 0..15;
  // lanes 16-31 = column lane-16, K=16..31.
  const int colh = lane & 15;
  const int ib   = (lane < 16) ? 0 : 16;
  v16h bfrag[2];
  #pragma unroll
  for (int nt = 0; nt < 2; ++nt) {
    const int bcol  = nt * 16 + colh;
    const float* xp = X + ((size_t)bcol * Kc + k) * Ic + ib;
    bfrag[nt] = cvt16(*(const float4*)(xp + 0), *(const float4*)(xp + 4),
                      *(const float4*)(xp + 8), *(const float4*)(xp + 12));
  }

  // A fragments (16x32 f16): lanes 0-15 = row `lane`, elems 0-15 = K0-15;
  // lanes 16-31 = row lane-16, K16-31.
  const float* Wk = W + (size_t)k * MROWS * Ic;
  const int rowh  = lane & 15;
  const int ia    = (lane < 16) ? 0 : 16;

  for (int mt = wv; mt < MROWS / 16; mt += 8) {
    const float* wp = Wk + (size_t)(mt * 16 + rowh) * Ic;
    v16h a = cvt16(*(const float4*)(wp + ia + 0),  *(const float4*)(wp + ia + 4),
                   *(const float4*)(wp + ia + 8), *(const float4*)(wp + ia + 12));
    #pragma unroll
    for (int nt = 0; nt < 2; ++nt) {
      v8f acc = {};
      acc = __builtin_amdgcn_wmma_f32_16x16x32_f16(
          /*neg_a=*/false, a, /*neg_b=*/false, bfrag[nt],
          /*c_mod=*/(short)0, acc, /*reuse_a=*/false, /*reuse_b=*/false);
      // D layout: lane elem j -> M = j + (lane<16?0:8), N = lane&15.
      // 8 consecutive r values per lane -> one 16B f16 store.
      const int bcol = nt * 16 + colh;
      const int r0   = mt * 16 + ((lane < 16) ? 0 : 8);
      v8h h;
      #pragma unroll
      for (int j = 0; j < 8; ++j) h[j] = (_Float16)acc[j];
      *(v8h*)(uhat + ((size_t)bcol * Kc + k) * MROWS + r0) = h;
    }
  }
}

// -------------------------------------------------------------------------
// Kernel 2: one fused routing pass over u_hat.
//   mode 0: c uniform (1/C), accumulate s only           (iteration 0)
//   mode 1: agr=u.v, b=agr (prev 0), write b, softmax, s (iteration 0->1)
//   mode 2: agr=u.v, b+=stored, softmax, s, write c_out  (iteration 1->2)
// Block = (b, 64-k chunk); each of 8 waves does 8 k's. Lane l owns classes
// l and l+32; softmax over C=64 via wave32 shuffle reduction. Partial s kept
// in 64 VGPRs/lane, merged via LDS atomics, then 1 global atomic / element.
// -------------------------------------------------------------------------
__global__ __launch_bounds__(256)
void caps_routing_pass(const _Float16* __restrict__ uhat,
                       const float* __restrict__ v_in,
                       float* __restrict__ bij,
                       float* __restrict__ s_out,
                       float* __restrict__ c_out,
                       int mode)
{
  __shared__ float v_lds[MROWS];
  __shared__ float s_lds[MROWS];

  const int b     = blockIdx.x >> 4;           // 16 k-chunks per b
  const int kbase = (blockIdx.x & 15) * 64;
  const int lane  = threadIdx.x & 31;
  const int wv    = threadIdx.x >> 5;
  const int tid   = threadIdx.x;

  #pragma unroll
  for (int j = 0; j < 8; ++j) {
    const int idx = tid * 8 + j;
    s_lds[idx] = 0.0f;
    v_lds[idx] = (mode != 0) ? v_in[(size_t)b * MROWS + idx] : 0.0f;
  }
  __syncthreads();

  const int c0 = lane;
  const int c1 = lane + 32;

  float sacc0[32], sacc1[32];
  #pragma unroll
  for (int o = 0; o < 32; ++o) { sacc0[o] = 0.0f; sacc1[o] = 0.0f; }

  for (int t = 0; t < 8; ++t) {
    const int k = kbase + wv * 8 + t;
    const _Float16* up = uhat + ((size_t)b * Kc + k) * MROWS;
    v8h u0[4], u1[4];
    #pragma unroll
    for (int q = 0; q < 4; ++q) {
      u0[q] = *(const v8h*)(up + c0 * 32 + q * 8);
      u1[q] = *(const v8h*)(up + c1 * 32 + q * 8);
    }

    float cw0, cw1;
    if (mode == 0) {
      cw0 = cw1 = 1.0f / 64.0f;
    } else {
      float agr0 = 0.0f, agr1 = 0.0f;
      #pragma unroll
      for (int q = 0; q < 4; ++q)
        #pragma unroll
        for (int j = 0; j < 8; ++j) {
          agr0 += (float)u0[q][j] * v_lds[c0 * 32 + q * 8 + j];
          agr1 += (float)u1[q][j] * v_lds[c1 * 32 + q * 8 + j];
        }
      const size_t bbase = ((size_t)b * Kc + k) * Cc;
      float bn0 = agr0, bn1 = agr1;
      if (mode == 2) { bn0 += bij[bbase + c0]; bn1 += bij[bbase + c1]; }
      else           { bij[bbase + c0] = bn0;  bij[bbase + c1] = bn1; }

      // softmax over the 64 class logits (2 per lane, wave32 reduction)
      float m = fmaxf(bn0, bn1);
      #pragma unroll
      for (int d = 16; d > 0; d >>= 1) m = fmaxf(m, __shfl_xor(m, d, 32));
      const float e0 = __expf(bn0 - m), e1 = __expf(bn1 - m);
      float sm = e0 + e1;
      #pragma unroll
      for (int d = 16; d > 0; d >>= 1) sm += __shfl_xor(sm, d, 32);
      const float inv = 1.0f / sm;
      cw0 = e0 * inv; cw1 = e1 * inv;
      if (mode == 2) { c_out[bbase + c0] = cw0; c_out[bbase + c1] = cw1; }
    }

    #pragma unroll
    for (int q = 0; q < 4; ++q)
      #pragma unroll
      for (int j = 0; j < 8; ++j) {
        sacc0[q * 8 + j] += cw0 * (float)u0[q][j];
        sacc1[q * 8 + j] += cw1 * (float)u1[q][j];
      }
  }

  #pragma unroll
  for (int o = 0; o < 32; ++o) {
    atomicAdd(&s_lds[c0 * 32 + o], sacc0[o]);   // ds_add_f32
    atomicAdd(&s_lds[c1 * 32 + o], sacc1[o]);
  }
  __syncthreads();
  #pragma unroll
  for (int j = 0; j < 8; ++j) {
    const int idx = tid * 8 + j;
    atomicAdd(&s_out[(size_t)b * MROWS + idx], s_lds[idx]);  // global_atomic_add_f32
  }
}

// -------------------------------------------------------------------------
// Kernel 3: squash over O=32. One (b,c) group per wave32.
// -------------------------------------------------------------------------
__global__ __launch_bounds__(256)
void caps_squash(const float* __restrict__ s, float* __restrict__ vout)
{
  const int gid = blockIdx.x * 256 + threadIdx.x;   // B*C*O = 65536 threads
  const float xv = s[gid];
  float sq = xv * xv;
  #pragma unroll
  for (int d = 16; d > 0; d >>= 1) sq += __shfl_xor(sq, d, 32);
  vout[gid] = (sq / (1.0f + sq)) * xv * rsqrtf(sq + 1e-8f);
}

// -------------------------------------------------------------------------
extern "C" void kernel_launch(void* const* d_in, const int* in_sizes, int n_in,
                              void* d_out, int out_size, void* d_ws, size_t ws_size,
                              hipStream_t stream)
{
  (void)in_sizes; (void)n_in; (void)out_size; (void)ws_size;
  const float* X = (const float*)d_in[0];   // [B,K,I]
  const float* W = (const float*)d_in[1];   // [K,C,O,I]

  float* vout = (float*)d_out;                        // [B,C,O]  = 65536 f32
  float* cout = vout + (size_t)Bsz * Cc * Oc;         // [B,K,C]  = 2097152 f32

  char* ws = (char*)d_ws;
  const size_t UHAT_BYTES = (size_t)Bsz * Kc * MROWS * sizeof(_Float16); // 128 MiB
  const size_t BIJ_BYTES  = (size_t)Bsz * Kc * Cc * sizeof(float);       //   8 MiB
  const size_t S_BYTES    = (size_t)Bsz * MROWS * sizeof(float);         // 256 KiB
  _Float16* uhat = (_Float16*)ws;
  float* bij  = (float*)(ws + UHAT_BYTES);
  float* sbuf = (float*)(ws + UHAT_BYTES + BIJ_BYTES);
  float* vbuf = (float*)(ws + UHAT_BYTES + BIJ_BYTES + S_BYTES);

  // u_hat einsum: 1024 x GEMM(2048x32x32) via v_wmma_f32_16x16x32_f16
  caps_uhat_wmma<<<Kc, 256, 0, stream>>>(W, X, uhat);

  // routing iteration 0 (uniform c)
  hipMemsetAsync(sbuf, 0, S_BYTES, stream);
  caps_routing_pass<<<Bsz * 16, 256, 0, stream>>>(uhat, vbuf, bij, sbuf, cout, 0);
  caps_squash<<<256, 256, 0, stream>>>(sbuf, vbuf);                    // v0

  // routing iteration 1 (b = agreement(v0))
  hipMemsetAsync(sbuf, 0, S_BYTES, stream);
  caps_routing_pass<<<Bsz * 16, 256, 0, stream>>>(uhat, vbuf, bij, sbuf, cout, 1);
  caps_squash<<<256, 256, 0, stream>>>(sbuf, vbuf);                    // v1

  // routing iteration 2 (b += agreement(v1); emit final c_ij and v_j)
  hipMemsetAsync(sbuf, 0, S_BYTES, stream);
  caps_routing_pass<<<Bsz * 16, 256, 0, stream>>>(uhat, vbuf, bij, sbuf, cout, 2);
  caps_squash<<<256, 256, 0, stream>>>(sbuf, vout);                    // v2 -> output
}
